// DiagonalAttention_50208167690563
// MI455X (gfx1250) — matrix operand order repaired and checked by reading, not compiled, not run on hardware
//
#include <hip/hip_runtime.h>

// ---------------- problem sizes (fixed by the reference) ----------------
static constexpr int Bn = 16;
static constexpr int Ls = 2048;
static constexpr int Ds = 1024;
static constexpr int Hs = 1024;
static constexpr float NEGV = -10000.0f;

typedef __attribute__((ext_vector_type(16))) __bf16 v16bf;
typedef __attribute__((ext_vector_type(8)))  float  v8f;

union ABf {
    v16bf v;
    __bf16 h[16];
    uint4  q[2];
};

__device__ __forceinline__ void cvt8(const float* __restrict__ p, __bf16* dst) {
    float4 f0 = ((const float4*)p)[0];
    float4 f1 = ((const float4*)p)[1];
    dst[0] = (__bf16)f0.x; dst[1] = (__bf16)f0.y;
    dst[2] = (__bf16)f0.z; dst[3] = (__bf16)f0.w;
    dst[4] = (__bf16)f1.x; dst[5] = (__bf16)f1.y;
    dst[6] = (__bf16)f1.z; dst[7] = (__bf16)f1.w;
}

// ======================================================================
// Kernel 0: one-shot W f32 -> bf16 (removes per-use conversions in proj)
// ======================================================================
__global__ __launch_bounds__(256) void cvtW_kernel(
    const float* __restrict__ W, __bf16* __restrict__ Wh)
{
    const size_t i = ((size_t)blockIdx.x * 256 + threadIdx.x) * 4;
    float4 f = *(const float4*)(W + i);
    __bf16 o[4] = { (__bf16)f.x, (__bf16)f.y, (__bf16)f.z, (__bf16)f.w };
    *(ulonglong1*)(Wh + i) = *(ulonglong1*)o;
}

// ======================================================================
// Kernel 1: q = relu(x1 Wh^T) * diag ; k = relu(x2 Wh^T)   (bf16 out)
//   grid (N/16, H/512, 2), block 256 (8 waves). Wave owns 16x64 C tile.
// ======================================================================
__global__ __launch_bounds__(256) void proj_kernel(
    const float* __restrict__ x1, const float* __restrict__ x2,
    const __bf16* __restrict__ Wh, const float* __restrict__ diag,
    __bf16* __restrict__ qh, __bf16* __restrict__ kh)
{
    const int lane = threadIdx.x & 31;
    const int wave = threadIdx.x >> 5;
    const int l15  = lane & 15;
    const int half = lane >> 4;

    const int  i0   = blockIdx.x * 16;
    const int  hw   = blockIdx.y * 512 + wave * 64;
    const bool is_q = (blockIdx.z == 0);
    const float* X   = is_q ? x1 : x2;
    __bf16*      out = is_q ? qh : kh;

    v8f acc[4] = {};

    const float* arow = X + (size_t)(i0 + l15) * Ds;

    for (int kd = 0; kd < Ds; kd += 32) {
        // A operand (16x32 bf16), ISA layout: lane<16 K={0..7,16..23}, lane>=16 K={8..15,24..31}
        ABf a;
        cvt8(arow + kd + half * 8,      &a.h[0]);
        cvt8(arow + kd + half * 8 + 16, &a.h[8]);
        #pragma unroll
        for (int t = 0; t < 4; ++t) {
            // B operand (32x16): lane<16 K={0..7,16..23} of column (lane&15), lane>=16 K={8..15,24..31}
            const __bf16* bp = Wh + (size_t)(hw + t * 16 + l15) * Ds + kd + half * 8;
            ABf bb;
            bb.q[0] = *(const uint4*)bp;
            bb.q[1] = *(const uint4*)(bp + 16);
            acc[t] = __builtin_amdgcn_wmma_f32_16x16x32_bf16(
                false, a.v, false, bb.v, (short)0, acc[t], false, false);
        }
    }

    // epilogue: relu (+diag for q), convert to bf16, store
    #pragma unroll
    for (int t = 0; t < 4; ++t) {
        const int hcol = hw + t * 16 + l15;
        const float dg = is_q ? diag[hcol] : 1.0f;
        #pragma unroll
        for (int r = 0; r < 8; ++r) {
            const int row = r + half * 8;
            float vv = acc[t][r];
            vv = vv > 0.0f ? vv : 0.0f;
            out[(size_t)(i0 + row) * Hs + hcol] = (__bf16)(vv * dg);
        }
    }
}

// ======================================================================
// Kernel 2: vt[b][d][l] = (bf16) x2[b][l][d]   (LDS-tiled transpose)
// ======================================================================
__global__ __launch_bounds__(256) void transpose_kernel(
    const float* __restrict__ x2, __bf16* __restrict__ vt)
{
    __shared__ __bf16 tile[32][33];
    const int b  = blockIdx.z;
    const int d0 = blockIdx.x * 32;
    const int l0 = blockIdx.y * 32;
    const int tx = threadIdx.x;   // 0..31
    const int ty = threadIdx.y;   // 0..7
    #pragma unroll
    for (int yy = ty; yy < 32; yy += 8)
        tile[yy][tx] = (__bf16)x2[((size_t)b * Ls + l0 + yy) * Ds + d0 + tx];
    __syncthreads();
    #pragma unroll
    for (int yy = ty; yy < 32; yy += 8)
        vt[((size_t)b * Ds + d0 + yy) * Ls + l0 + tx] = tile[tx][yy];
}

// ======================================================================
// Kernel 3: flash attention.  One block = (b, 16 i-rows), 8 waves.
//   Wave w: H-slice [128w,128w+128) for S partials; d-slice [128w,128w+128)
//   for the output accumulator (8 WMMA C tiles, f32).
// ======================================================================
__global__ __launch_bounds__(256) void attn_kernel(
    const __bf16* __restrict__ qh, const __bf16* __restrict__ kh,
    const __bf16* __restrict__ vt, const long long* __restrict__ mask,
    float* __restrict__ out)
{
    __shared__ __align__(16) float  S[16 * 32];
    __shared__ __align__(32) __bf16 P[16 * 32];
    __shared__ float scaleS[16];
    __shared__ float linv[16];

    const int b    = blockIdx.y;
    const int i0   = blockIdx.x * 16;
    const int lane = threadIdx.x & 31;
    const int wave = threadIdx.x >> 5;
    const int l15  = lane & 15;
    const int half = lane >> 4;
    const int hs   = wave * 128;   // H slice for score partials
    const int dw   = wave * 128;   // D slice for output accumulation

    v8f acc[8] = {};
    float m_r = -3.0e38f, l_r = 0.0f;   // live in wave0 lanes 0..15

    const __bf16* qrow = qh + ((size_t)b * Ls + i0 + l15) * Hs;

    for (int j0 = 0; j0 < Ls; j0 += 32) {
        __syncthreads();
        S[threadIdx.x] = 0.0f;
        S[threadIdx.x + 256] = 0.0f;
        __syncthreads();

        // ---- partial S = q . k^T over h in [hs, hs+128) ----
        #pragma unroll
        for (int jt = 0; jt < 2; ++jt) {
            v8f c = {};
            const __bf16* krow = kh + ((size_t)b * Ls + j0 + jt * 16 + l15) * Hs;
            #pragma unroll
            for (int kk = 0; kk < 4; ++kk) {
                const int h0 = hs + kk * 32;
                ABf a, bb;
                const __bf16* ap = qrow + h0 + half * 8;
                a.q[0] = *(const uint4*)ap;
                a.q[1] = *(const uint4*)(ap + 16);
                const __bf16* bp = krow + h0 + half * 8;
                bb.q[0] = *(const uint4*)bp;
                bb.q[1] = *(const uint4*)(bp + 16);
                c = __builtin_amdgcn_wmma_f32_16x16x32_bf16(
                    false, a.v, false, bb.v, (short)0, c, false, false);
            }
            #pragma unroll
            for (int r = 0; r < 8; ++r)
                atomicAdd(&S[(r + half * 8) * 32 + jt * 16 + l15], c[r]); // ds_add_f32
        }
        __syncthreads();

        // ---- online softmax for the 16 rows (wave0 lanes 0..15) ----
        if (threadIdx.x < 16) {
            const int r = threadIdx.x;
            float sv[32];
            #pragma unroll
            for (int j = 0; j < 32; ++j) {
                const float mk = (float)mask[(size_t)b * Ls + j0 + j];
                sv[j] = S[r * 32 + j] + (1.0f - mk) * NEGV;
            }
            float mnew = m_r;
            #pragma unroll
            for (int j = 0; j < 32; ++j) mnew = fmaxf(mnew, sv[j]);
            const float sc = __expf(m_r - mnew);
            float sum = 0.0f;
            #pragma unroll
            for (int j = 0; j < 32; ++j) {
                const float p = __expf(sv[j] - mnew);
                sum += p;
                P[r * 32 + j] = (__bf16)p;
            }
            l_r = l_r * sc + sum;
            m_r = mnew;
            scaleS[r] = sc;
        }
        __syncthreads();

        // ---- rescale accumulators by exp(m_old - m_new) ----
        float sc[8];
        #pragma unroll
        for (int r = 0; r < 8; ++r) sc[r] = scaleS[r + half * 8];
        #pragma unroll
        for (int t = 0; t < 8; ++t)
            #pragma unroll
            for (int r = 0; r < 8; ++r) acc[t][r] *= sc[r];

        // ---- acc += P (16x32) x V (32x16 per d-subtile) ----
        ABf ap;
        {
            const __bf16* pp = &P[l15 * 32 + half * 8];
            ap.q[0] = *(const uint4*)pp;
            ap.q[1] = *(const uint4*)(pp + 16);
        }
        #pragma unroll
        for (int t = 0; t < 8; ++t) {
            const __bf16* vp = vt + ((size_t)b * Ds + dw + t * 16 + l15) * Ls
                                  + j0 + half * 8;
            ABf bb;
            bb.q[0] = *(const uint4*)vp;
            bb.q[1] = *(const uint4*)(vp + 16);
            acc[t] = __builtin_amdgcn_wmma_f32_16x16x32_bf16(
                false, ap.v, false, bb.v, (short)0, acc[t], false, false);
        }
    }

    __syncthreads();
    if (threadIdx.x < 16) linv[threadIdx.x] = (l_r > 0.0f) ? (1.0f / l_r) : 0.0f;
    __syncthreads();

    float li[8];
    #pragma unroll
    for (int r = 0; r < 8; ++r) li[r] = linv[r + half * 8];
    #pragma unroll
    for (int t = 0; t < 8; ++t) {
        const int d = dw + t * 16 + l15;
        #pragma unroll
        for (int r = 0; r < 8; ++r)
            out[((size_t)b * Ls + i0 + r + half * 8) * Ds + d] = acc[t][r] * li[r];
    }
}

// ======================================================================
extern "C" void kernel_launch(void* const* d_in, const int* in_sizes, int n_in,
                              void* d_out, int out_size, void* d_ws, size_t ws_size,
                              hipStream_t stream) {
    const float*     x1   = (const float*)d_in[0];
    const float*     x2   = (const float*)d_in[1];
    const long long* mask = (const long long*)d_in[2];  // int64 ones
    const float*     W    = (const float*)d_in[3];
    const float*     diag = (const float*)d_in[4];
    float*           out  = (float*)d_out;

    char* ws = (char*)d_ws;
    const size_t seg = (size_t)Bn * Ls * Hs * sizeof(__bf16);  // 64 MB each
    __bf16* qh = (__bf16*)(ws);
    __bf16* kh = (__bf16*)(ws + seg);
    __bf16* vt = (__bf16*)(ws + 2 * seg);
    __bf16* Wh = (__bf16*)(ws + 3 * seg);                      // +2 MB

    // 0) W -> bf16 once (1M elements)
    cvtW_kernel<<<dim3((Hs * Ds) / (256 * 4)), 256, 0, stream>>>(W, Wh);
    // 1) projections: q = relu(x1 W^T)*diag, k = relu(x2 W^T)
    proj_kernel<<<dim3((Bn * Ls) / 16, Hs / 512, 2), 256, 0, stream>>>(
        x1, x2, Wh, diag, qh, kh);
    // 2) v^T in bf16 for the P.V matmul B-operand
    transpose_kernel<<<dim3(Ds / 32, Ls / 32, Bn), dim3(32, 8), 0, stream>>>(x2, vt);
    // 3) fused masked-softmax attention
    attn_kernel<<<dim3(Ls / 16, Bn), 256, 0, stream>>>(qh, kh, vt, mask, out);
}
